// GateLayer_61821759258647
// MI455X (gfx1250) — matrix operand order repaired and checked by reading, not compiled, not run on hardware
//
#include <hip/hip_runtime.h>
#include <hip/hip_bf16.h>

// ---------------------------------------------------------------------------
// MoE gating layer for MI455X (gfx1250), wave32 + WMMA bf16.
//   B=8192, IN_DIM=4096, HID=2048, E=64, TOP_K=8, THRESHOLD=0
// ---------------------------------------------------------------------------

#define USE_ASYNC 1

#define Bdim    8192
#define IN_DIM  4096
#define HID     2048
#define NEXP    64
#define TOPK    8

typedef __attribute__((ext_vector_type(16))) __bf16 v16bf;
typedef __attribute__((ext_vector_type(8)))  float  v8f;

static_assert(sizeof(v16bf) == 32, "v16bf must be 32 bytes");
static_assert(sizeof(v8f) == 32, "v8f must be 32 bytes");

union FragAB { v16bf v; uint4 q[2]; };

__device__ __forceinline__ v8f v8f_zero() {
  v8f z = {0.f, 0.f, 0.f, 0.f, 0.f, 0.f, 0.f, 0.f};
  return z;
}

// ---- global -> LDS 16B mover: async (ASYNCcnt-tracked) or sync fallback ----
__device__ __forceinline__ void g2l_b128(void* lds, const void* g) {
#if USE_ASYNC
  asm volatile("global_load_async_to_lds_b128 %0, %1, off"
               :: "v"((unsigned)(unsigned long long)lds), "v"(g)
               : "memory");
#else
  *(uint4*)lds = *(const uint4*)g;
#endif
}
__device__ __forceinline__ void g2l_commit() {
#if USE_ASYNC
  asm volatile("s_wait_asynccnt 0x0" ::: "memory");
#endif
}

// convert 16 f32 -> 16 bf16, store contiguous into LDS (16B-aligned dest)
__device__ __forceinline__ void cvt_store16(__bf16* d, float4 a0, float4 a1,
                                            float4 a2, float4 a3) {
  union { __bf16 h[16]; uint4 q[2]; } u;
  u.h[0]  = (__bf16)a0.x; u.h[1]  = (__bf16)a0.y; u.h[2]  = (__bf16)a0.z; u.h[3]  = (__bf16)a0.w;
  u.h[4]  = (__bf16)a1.x; u.h[5]  = (__bf16)a1.y; u.h[6]  = (__bf16)a1.z; u.h[7]  = (__bf16)a1.w;
  u.h[8]  = (__bf16)a2.x; u.h[9]  = (__bf16)a2.y; u.h[10] = (__bf16)a2.z; u.h[11] = (__bf16)a2.w;
  u.h[12] = (__bf16)a3.x; u.h[13] = (__bf16)a3.y; u.h[14] = (__bf16)a3.z; u.h[15] = (__bf16)a3.w;
  *(uint4*)d       = u.q[0];
  *(uint4*)(d + 8) = u.q[1];
}

// ---------------------------------------------------------------------------
// Kernel: f32 -> bf16 elementwise (8 elems/thread)
// ---------------------------------------------------------------------------
__global__ void cvt_bf16_kernel(const float* __restrict__ src,
                                __bf16* __restrict__ dst, int n8) {
  int i = blockIdx.x * blockDim.x + threadIdx.x;
  if (i < n8) {
    const float4* s = (const float4*)src + (size_t)i * 2;
    float4 a = s[0], b = s[1];
    union { __bf16 h[8]; uint4 q; } u;
    u.h[0] = (__bf16)a.x; u.h[1] = (__bf16)a.y; u.h[2] = (__bf16)a.z; u.h[3] = (__bf16)a.w;
    u.h[4] = (__bf16)b.x; u.h[5] = (__bf16)b.y; u.h[6] = (__bf16)b.z; u.h[7] = (__bf16)b.w;
    ((uint4*)dst)[i] = u.q;
  }
}

// ---------------------------------------------------------------------------
// GEMM1: h = relu(x @ W1 + b1)   [8192,4096] x [4096,2048] -> bf16 [8192,2048]
// Block: 256 thr (8 waves), tile 128(M) x 128(N), K-step 32, double-buffered.
// ---------------------------------------------------------------------------
#define LDA 40    // halves; 20 dwords -> conflict-free b128 fragment loads
#define LDB 136   // halves; 68 dwords

__global__ __launch_bounds__(256, 1)
void gemm1_kernel(const float* __restrict__ X, const __bf16* __restrict__ W1b,
                  const float* __restrict__ b1, __bf16* __restrict__ Hb) {
  __shared__ __bf16 sA[2][128 * LDA];
  __shared__ __bf16 sB[2][32 * LDB];

  const int tid = threadIdx.x;
  const int lane = tid & 31, wave = tid >> 5;
  const int m0 = blockIdx.y * 128, n0 = blockIdx.x * 128;
  const int wm = wave & 3, wn = wave >> 2;
  const int mbase = wm * 32, nbase = wn * 64;

  v8f acc[2][4];
#pragma unroll
  for (int i = 0; i < 2; ++i)
#pragma unroll
    for (int j = 0; j < 4; ++j) acc[i][j] = v8f_zero();

  // global->LDS assignments
  const int ar = tid >> 1, acc16 = (tid & 1) * 16;   // A: 16 f32 per thread
  const int br = tid >> 3, bc = (tid & 7) * 16;      // B: 16 bf16 per thread

  const float4* gA = (const float4*)(X + (size_t)(m0 + ar) * IN_DIM + acc16);
  const __bf16* gB = W1b + (size_t)br * HID + n0 + bc;

  // fragment addressing (ISA 7.12.2): A lane->(M, koff), B lane->(N, koff)
  const int arow = lane & 15, koff = (lane >> 4) * 8;

  // ---- prologue: fill buffer 0
  {
    float4 a0 = gA[0], a1 = gA[1], a2 = gA[2], a3 = gA[3];
    cvt_store16(&sA[0][ar * LDA + acc16], a0, a1, a2, a3);
    g2l_b128(&sB[0][br * LDB + bc], gB);
    g2l_b128(&sB[0][br * LDB + bc + 8], gB + 8);
  }
  g2l_commit();
  __syncthreads();

  const int NK = IN_DIM / 32;  // 128
  for (int ks = 0; ks < NK; ++ks) {
    const int cur = ks & 1, nxt = cur ^ 1;
    const bool pre = (ks + 1) < NK;
    float4 a0, a1, a2, a3;
    if (pre) {
      const float4* pg = gA + (size_t)(ks + 1) * 8;          // +32 floats
      a0 = pg[0]; a1 = pg[1]; a2 = pg[2]; a3 = pg[3];
      const __bf16* pb = gB + (size_t)(ks + 1) * 32 * HID;   // +32 K rows
      g2l_b128(&sB[nxt][br * LDB + bc], pb);
      g2l_b128(&sB[nxt][br * LDB + bc + 8], pb + 8);
    }
    // load fragments
    FragAB afr[2], bfr[4];
#pragma unroll
    for (int mt = 0; mt < 2; ++mt) {
      const __bf16* p = &sA[cur][(mbase + mt * 16 + arow) * LDA + koff];
      afr[mt].q[0] = *(const uint4*)p;
      afr[mt].q[1] = *(const uint4*)(p + 16);
    }
#pragma unroll
    for (int nt = 0; nt < 4; ++nt) {
      const __bf16* p = &sB[cur][nbase + nt * 16 + (lane & 15)];
#pragma unroll
      for (int s = 0; s < 16; ++s)
        bfr[nt].v[s] = p[((s & 8) * 2 + koff + (s & 7)) * LDB];
    }
#pragma unroll
    for (int mt = 0; mt < 2; ++mt)
#pragma unroll
      for (int nt = 0; nt < 4; ++nt)
        acc[mt][nt] = __builtin_amdgcn_wmma_f32_16x16x32_bf16(
            false, afr[mt].v, false, bfr[nt].v, (short)0, acc[mt][nt], false, false);
    if (pre) {
      cvt_store16(&sA[nxt][ar * LDA + acc16], a0, a1, a2, a3);
      g2l_commit();
    }
    __syncthreads();
  }

  // ---- epilogue: bias + relu -> bf16
#pragma unroll
  for (int mt = 0; mt < 2; ++mt) {
#pragma unroll
    for (int nt = 0; nt < 4; ++nt) {
      const int col = n0 + nbase + nt * 16 + (lane & 15);
      const float bias = b1[col];
#pragma unroll
      for (int r = 0; r < 8; ++r) {
        const int row = m0 + mbase + mt * 16 + r + 8 * (lane >> 4);
        float v = acc[mt][nt][r] + bias;
        v = v > 0.f ? v : 0.f;
        Hb[(size_t)row * HID + col] = (__bf16)v;
      }
    }
  }
}

// ---------------------------------------------------------------------------
// GEMM2: logits = h @ W2 + b2 + eps * (x @ nw)     -> f32 [8192, 64]
// Block: 256 thr (8 waves), tile 128(M) x 64(N); two K phases.
// ---------------------------------------------------------------------------
__global__ __launch_bounds__(256, 1)
void gemm2_kernel(const __bf16* __restrict__ Hb, const __bf16* __restrict__ W2b,
                  const float* __restrict__ b2, const float* __restrict__ X,
                  const __bf16* __restrict__ NWb, const float* __restrict__ eps,
                  float* __restrict__ logits) {
  __shared__ __bf16 sA[2][128 * LDA];
  __shared__ __bf16 sB[2][32 * LDB];

  const int tid = threadIdx.x;
  const int lane = tid & 31, wave = tid >> 5;
  const int m0 = blockIdx.x * 128;

  v8f acc1[4], acc2[4];
#pragma unroll
  for (int j = 0; j < 4; ++j) { acc1[j] = v8f_zero(); acc2[j] = v8f_zero(); }

  const int ar = tid >> 1, ach = (tid & 1) * 16;  // A: 16 elems per thread
  const int br = tid >> 3, bch = (tid & 7) * 8;   // B: 8 halves (16B) per thread
  const int arow = lane & 15, koff = (lane >> 4) * 8;

  // ============ phase 1: K over HID, A = Hb (bf16), B = W2b ============
  {
    const __bf16* gA = Hb + (size_t)(m0 + ar) * HID + ach;
    const __bf16* gB = W2b + (size_t)br * NEXP + bch;
    g2l_b128(&sA[0][ar * LDA + ach], gA);
    g2l_b128(&sA[0][ar * LDA + ach + 8], gA + 8);
    g2l_b128(&sB[0][br * LDB + bch], gB);
    g2l_commit();
    __syncthreads();

    const int NK = HID / 32;  // 64
    for (int ks = 0; ks < NK; ++ks) {
      const int cur = ks & 1, nxt = cur ^ 1;
      const bool pre = (ks + 1) < NK;
      if (pre) {
        const __bf16* pa = gA + (size_t)(ks + 1) * 32;
        const __bf16* pb = gB + (size_t)(ks + 1) * 32 * NEXP;
        g2l_b128(&sA[nxt][ar * LDA + ach], pa);
        g2l_b128(&sA[nxt][ar * LDA + ach + 8], pa + 8);
        g2l_b128(&sB[nxt][br * LDB + bch], pb);
      }
      FragAB af, bf[4];
      const __bf16* p = &sA[cur][(wave * 16 + arow) * LDA + koff];
      af.q[0] = *(const uint4*)p;
      af.q[1] = *(const uint4*)(p + 16);
#pragma unroll
      for (int nt = 0; nt < 4; ++nt) {
        const __bf16* pb = &sB[cur][nt * 16 + (lane & 15)];
#pragma unroll
        for (int s = 0; s < 16; ++s)
          bf[nt].v[s] = pb[((s & 8) * 2 + koff + (s & 7)) * LDB];
      }
#pragma unroll
      for (int nt = 0; nt < 4; ++nt)
        acc1[nt] = __builtin_amdgcn_wmma_f32_16x16x32_bf16(
            false, af.v, false, bf[nt].v, (short)0, acc1[nt], false, false);
      if (pre) g2l_commit();
      __syncthreads();
    }
  }

  // ============ phase 2: K over IN_DIM, A = X (f32->bf16), B = NWb ============
  {
    const float4* gA = (const float4*)(X + (size_t)(m0 + ar) * IN_DIM + ach);
    const __bf16* gB = NWb + (size_t)br * NEXP + bch;
    {
      float4 a0 = gA[0], a1 = gA[1], a2 = gA[2], a3 = gA[3];
      cvt_store16(&sA[0][ar * LDA + ach], a0, a1, a2, a3);
      g2l_b128(&sB[0][br * LDB + bch], gB);
      g2l_commit();
    }
    __syncthreads();

    const int NK = IN_DIM / 32;  // 128
    for (int ks = 0; ks < NK; ++ks) {
      const int cur = ks & 1, nxt = cur ^ 1;
      const bool pre = (ks + 1) < NK;
      float4 a0, a1, a2, a3;
      if (pre) {
        const float4* pg = gA + (size_t)(ks + 1) * 8;
        a0 = pg[0]; a1 = pg[1]; a2 = pg[2]; a3 = pg[3];
        g2l_b128(&sB[nxt][br * LDB + bch], gB + (size_t)(ks + 1) * 32 * NEXP);
      }
      FragAB af, bf[4];
      const __bf16* p = &sA[cur][(wave * 16 + arow) * LDA + koff];
      af.q[0] = *(const uint4*)p;
      af.q[1] = *(const uint4*)(p + 16);
#pragma unroll
      for (int nt = 0; nt < 4; ++nt) {
        const __bf16* pb = &sB[cur][nt * 16 + (lane & 15)];
#pragma unroll
        for (int s = 0; s < 16; ++s)
          bf[nt].v[s] = pb[((s & 8) * 2 + koff + (s & 7)) * LDB];
      }
#pragma unroll
      for (int nt = 0; nt < 4; ++nt)
        acc2[nt] = __builtin_amdgcn_wmma_f32_16x16x32_bf16(
            false, af.v, false, bf[nt].v, (short)0, acc2[nt], false, false);
      if (pre) {
        cvt_store16(&sA[nxt][ar * LDA + ach], a0, a1, a2, a3);
        g2l_commit();
      }
      __syncthreads();
    }
  }

  // ---- epilogue: logits = acc1 + b2 + eps * acc2
#pragma unroll
  for (int nt = 0; nt < 4; ++nt) {
    const int col = nt * 16 + (lane & 15);
    const float bias = b2[col];
#pragma unroll
    for (int r = 0; r < 8; ++r) {
      const int row = m0 + wave * 16 + r + 8 * (lane >> 4);
      const size_t o = (size_t)row * NEXP + col;
      logits[o] = acc1[nt][r] + bias + eps[o] * acc2[nt][r];
    }
  }
}

// ---------------------------------------------------------------------------
// Softmax per row (64 experts, one wave per row) + per-block column partials
// ---------------------------------------------------------------------------
__global__ __launch_bounds__(256)
void softmax_colsum_kernel(const float* __restrict__ logits,
                           float* __restrict__ ew, float* __restrict__ partials) {
  __shared__ float tile[8][NEXP];
  const int lane = threadIdx.x & 31, wave = threadIdx.x >> 5;
  const int row = blockIdx.x * 8 + wave;

  float2 v = *(const float2*)(logits + (size_t)row * NEXP + lane * 2);
  float m = fmaxf(v.x, v.y);
#pragma unroll
  for (int off = 16; off; off >>= 1) m = fmaxf(m, __shfl_xor(m, off, 32));
  float e0 = expf(v.x - m), e1 = expf(v.y - m);
  float s = e0 + e1;
#pragma unroll
  for (int off = 16; off; off >>= 1) s += __shfl_xor(s, off, 32);
  const float inv = 1.0f / s;
  e0 *= inv; e1 *= inv;

  float2 r; r.x = e0; r.y = e1;
  *(float2*)(ew + (size_t)row * NEXP + lane * 2) = r;

  tile[wave][lane * 2] = e0;
  tile[wave][lane * 2 + 1] = e1;
  __syncthreads();
  if (threadIdx.x < NEXP) {
    float acc = 0.f;
#pragma unroll
    for (int t = 0; t < 8; ++t) acc += tile[t][threadIdx.x];
    partials[(size_t)blockIdx.x * NEXP + threadIdx.x] = acc;
  }
}

// ---------------------------------------------------------------------------
// Deterministic column-sum reduce + load-balance mask (1 block, 64 threads)
// ---------------------------------------------------------------------------
__global__ void stats_kernel(const float* __restrict__ partials,
                             const float* __restrict__ total_prev,
                             float* __restrict__ mask, int nblocks) {
  const int e = threadIdx.x;  // 64 threads = 2 waves
  float s = total_prev[e];
  for (int b = 0; b < nblocks; ++b) s += partials[(size_t)b * NEXP + e];

  __shared__ float sw[2];
  float t = s;
#pragma unroll
  for (int off = 16; off; off >>= 1) t += __shfl_xor(t, off, 32);
  if ((threadIdx.x & 31) == 0) sw[threadIdx.x >> 5] = t;
  __syncthreads();
  const float mean = (sw[0] + sw[1]) * (1.0f / (float)NEXP);
  mask[e] = ((s - mean) > 0.0f) ? 0.0f : 1.0f;
}

// ---------------------------------------------------------------------------
// Top-8 select per row + softmax over kept probability values (one wave/row)
// ---------------------------------------------------------------------------
__global__ __launch_bounds__(256)
void topk_kernel(const float* __restrict__ ew, const float* __restrict__ mask,
                 float* __restrict__ out) {
  const int lane = threadIdx.x & 31, wave = threadIdx.x >> 5;
  const int row = blockIdx.x * 8 + wave;
  const int e0i = lane * 2, e1i = lane * 2 + 1;

  const float v0 = ew[(size_t)row * NEXP + e0i] * mask[e0i];
  const float v1 = ew[(size_t)row * NEXP + e1i] * mask[e1i];

  // key = (float bits << 7) | (64 - idx): values are >= 0, so bit order is
  // monotone; ties resolve to the lowest index. key==0 marks "removed".
  unsigned long long k0 =
      ((unsigned long long)__float_as_uint(v0) << 7) | (unsigned)(64 - e0i);
  unsigned long long k1 =
      ((unsigned long long)__float_as_uint(v1) << 7) | (unsigned)(64 - e1i);
  bool keep0 = false, keep1 = false;

#pragma unroll
  for (int it = 0; it < TOPK; ++it) {
    unsigned long long kk = k0 > k1 ? k0 : k1;
#pragma unroll
    for (int off = 16; off; off >>= 1) {
      unsigned long long o = __shfl_xor(kk, off, 32);
      if (o > kk) kk = o;
    }
    const int idx = 64 - (int)(kk & 127ull);
    if (idx == e0i) { keep0 = true; k0 = 0ull; }
    if (idx == e1i) { keep1 = true; k1 = 0ull; }
  }

  float m = -3.4e38f;
  if (keep0) m = v0;
  if (keep1) m = fmaxf(m, v1);
#pragma unroll
  for (int off = 16; off; off >>= 1) m = fmaxf(m, __shfl_xor(m, off, 32));
  float s0 = keep0 ? expf(v0 - m) : 0.f;
  float s1 = keep1 ? expf(v1 - m) : 0.f;
  float s = s0 + s1;
#pragma unroll
  for (int off = 16; off; off >>= 1) s += __shfl_xor(s, off, 32);
  const float inv = 1.0f / s;

  float2 r; r.x = s0 * inv; r.y = s1 * inv;
  *(float2*)(out + (size_t)row * NEXP + lane * 2) = r;
}

// ---------------------------------------------------------------------------
// Host-side launcher
// ---------------------------------------------------------------------------
extern "C" void kernel_launch(void* const* d_in, const int* in_sizes, int n_in,
                              void* d_out, int out_size, void* d_ws, size_t ws_size,
                              hipStream_t stream) {
  (void)in_sizes; (void)n_in; (void)out_size; (void)ws_size;
  const float* x   = (const float*)d_in[0];
  const float* W1  = (const float*)d_in[1];
  const float* b1  = (const float*)d_in[2];
  const float* W2  = (const float*)d_in[3];
  const float* b2  = (const float*)d_in[4];
  const float* nw  = (const float*)d_in[5];
  const float* eps = (const float*)d_in[6];
  const float* tp  = (const float*)d_in[7];
  float* out = (float*)d_out;

  char* ws = (char*)d_ws;
  __bf16* W1b = (__bf16*)ws;  ws += (size_t)IN_DIM * HID * 2;   // 16.8 MB
  __bf16* W2b = (__bf16*)ws;  ws += (size_t)HID * NEXP * 2;     // 256 KB
  __bf16* NWb = (__bf16*)ws;  ws += (size_t)IN_DIM * NEXP * 2;  // 512 KB
  __bf16* Hb  = (__bf16*)ws;  ws += (size_t)Bdim * HID * 2;     // 33.6 MB
  float* logits   = (float*)ws; ws += (size_t)Bdim * NEXP * 4;  // 2 MB
  float* ewbuf    = (float*)ws; ws += (size_t)Bdim * NEXP * 4;  // 2 MB
  float* partials = (float*)ws; ws += (size_t)(Bdim / 8) * NEXP * 4;
  float* maskbuf  = (float*)ws;

  // 1) weight conversions to bf16 (row-major kept; WMMA B wants K-major rows)
  {
    int n8 = IN_DIM * HID / 8;
    cvt_bf16_kernel<<<(n8 + 255) / 256, 256, 0, stream>>>(W1, W1b, n8);
    n8 = HID * NEXP / 8;
    cvt_bf16_kernel<<<(n8 + 255) / 256, 256, 0, stream>>>(W2, W2b, n8);
    n8 = IN_DIM * NEXP / 8;
    cvt_bf16_kernel<<<(n8 + 255) / 256, 256, 0, stream>>>(nw, NWb, n8);
  }

  // 2) h = relu(x @ W1 + b1)
  gemm1_kernel<<<dim3(HID / 128, Bdim / 128), 256, 0, stream>>>(x, W1b, b1, Hb);

  // 3) logits = h @ W2 + b2 + eps * (x @ nw)
  gemm2_kernel<<<Bdim / 128, 256, 0, stream>>>(Hb, W2b, b2, x, NWb, eps, logits);

  // 4) rowwise softmax + deterministic column partials
  softmax_colsum_kernel<<<Bdim / 8, 256, 0, stream>>>(logits, ewbuf, partials);

  // 5) total = total_prev + colsum; mask = (total - mean > 0) ? 0 : 1
  stats_kernel<<<1, 64, 0, stream>>>(partials, tp, maskbuf, Bdim / 8);

  // 6) top-8 keep + softmax over kept values
  topk_kernel<<<Bdim / 8, 256, 0, stream>>>(ewbuf, maskbuf, out);
}
